// BigBirdGNN_67396626809057
// MI455X (gfx1250) — matrix operand order repaired and checked by reading, not compiled, not run on hardware
//
#include <hip/hip_runtime.h>

// ---------------------------------------------------------------------------
// BigBird GNN forward for gfx1250 (MI455X).
// GEMMs (h@Wq/Wk/Wv, attn@Wo) run on v_wmma_f32_16x16x32_f16 (f16 in, f32 acc).
// Edge softmax / scatter in fp32 with atomics (all buffers L2-resident).
// ---------------------------------------------------------------------------

typedef _Float16 v16h __attribute__((ext_vector_type(16)));
typedef _Float16 v8h  __attribute__((ext_vector_type(8)));
typedef _Float16 v4h  __attribute__((ext_vector_type(4)));
typedef float    v8f  __attribute__((ext_vector_type(8)));
typedef float    v4f  __attribute__((ext_vector_type(4)));

static constexpr int kD  = 128;          // feature dim
static constexpr int kDD = kD * kD;      // one weight matrix

// ---------------- threefry-2x32 (matches JAX) ------------------------------
__host__ __device__ __forceinline__ unsigned rotl32_(unsigned x, int r) {
  return (x << r) | (x >> (32 - r));
}

__host__ __device__ __forceinline__ void tf2x32(unsigned k0, unsigned k1,
                                                unsigned c0, unsigned c1,
                                                unsigned* o0, unsigned* o1) {
  const unsigned ks2 = k0 ^ k1 ^ 0x1BD11BDAu;
  unsigned ks[3] = {k0, k1, ks2};
  const int R0[4] = {13, 15, 26, 6};
  const int R1[4] = {17, 29, 16, 24};
  unsigned x0 = c0 + k0;
  unsigned x1 = c1 + k1;
  for (int blk = 0; blk < 5; ++blk) {
    const int* r = (blk & 1) ? R1 : R0;
    for (int i = 0; i < 4; ++i) {
      x0 += x1;
      x1 = rotl32_(x1, r[i]);
      x1 ^= x0;
    }
    x0 += ks[(blk + 1) % 3];
    x1 += ks[(blk + 2) % 3] + (unsigned)(blk + 1);
  }
  *o0 = x0;
  *o1 = x1;
}

// monotonic float <-> uint encoding for atomicMax-based segment max
__device__ __forceinline__ unsigned fenc(float f) {
  unsigned b = __float_as_uint(f);
  return (b & 0x80000000u) ? ~b : (b | 0x80000000u);
}
__device__ __forceinline__ float fdec(unsigned u) {
  return (u & 0x80000000u) ? __uint_as_float(u ^ 0x80000000u)
                           : __uint_as_float(~u);
}

// ---------------- weight pack: f32 [l][k][n] -> f16 B-fragment layout ------
// Wp[l*DD + (k>>4)*(D*16) + n*16 + (k&15)]  => each wave's B chunk is one
// contiguous 32B read of 16 halves (column n, K-chunk of 16).
__global__ void k_pack_w(const float* __restrict__ W, _Float16* __restrict__ Wp,
                         int total) {
  int gid = blockIdx.x * blockDim.x + threadIdx.x;
  if (gid >= total) return;
  int l  = gid / kDD;
  int r  = gid - l * kDD;
  int kk = r / kD;
  int n  = r - kk * kD;
  Wp[(size_t)l * kDD + (size_t)(kk >> 4) * (kD * 16) + (size_t)n * 16 +
     (kk & 15)] = (_Float16)W[gid];
}

// ---------------- f32 -> f16 (vector x4) -----------------------------------
__global__ void k_f32_to_f16(const float* __restrict__ in,
                             _Float16* __restrict__ out, int n4) {
  int gid = blockIdx.x * blockDim.x + threadIdx.x;
  if (gid >= n4) return;
  v4f v = *(const v4f*)(in + (size_t)gid * 4);
  v4h o;
  o[0] = (_Float16)v[0];
  o[1] = (_Float16)v[1];
  o[2] = (_Float16)v[2];
  o[3] = (_Float16)v[3];
  *(v4h*)(out + (size_t)gid * 4) = o;
}

// ---------------- zero f32 (vector x4) -------------------------------------
__global__ void k_zero_f32(float* __restrict__ p, int n4) {
  int gid = blockIdx.x * blockDim.x + threadIdx.x;
  if (gid >= n4) return;
  v4f z = {};
  *(v4f*)(p + (size_t)gid * 4) = z;
}

// ---------------- init segment max (enc(-inf)) and segment sum -------------
__global__ void k_init_ms(unsigned* __restrict__ mEnc, float* __restrict__ s,
                          int n) {
  int gid = blockIdx.x * blockDim.x + threadIdx.x;
  if (gid >= n) return;
  mEnc[gid] = 0x007FFFFFu;  // fenc(-inf)
  s[gid] = 0.0f;
}

// ---------------- WMMA GEMM: C[nrows x 128] = A_f16 @ Wp_f16 + bias --------
// block = 256 threads = 8 waves; wave w owns the 16-col tile nbase = 16*w.
// K=128 consumed by 4 x v_wmma_f32_16x16x32_f16 per wave (fp32 accumulate).
__global__ __launch_bounds__(256) void k_gemm(const _Float16* __restrict__ A,
                                              const _Float16* __restrict__ Wp,
                                              const float* __restrict__ bias,
                                              float* __restrict__ C, int nrows,
                                              int relu) {
  const int lane  = threadIdx.x & 31;
  const int wave  = threadIdx.x >> 5;  // 0..7 -> n-tile
  const int mbase = blockIdx.x * 16;
  const int nbase = wave * 16;
  if (mbase >= nrows) return;
  const int grp = lane >> 4;   // half-wave group
  const int nl  = lane & 15;

  const _Float16* arow = A + (size_t)(mbase + nl) * kD;
  v8f acc = {};
#pragma unroll
  for (int kt = 0; kt < 4; ++kt) {
    const int kbase = kt * 32;
    // A fragment (contiguous K, matching B): lanes 0-15 hold K 0..15,
    // lanes 16-31 hold K 16..31 -> two 16B row reads per lane.
    v8h alo = *(const v8h*)(arow + kbase + grp * 16);
    v8h ahi = *(const v8h*)(arow + kbase + grp * 16 + 8);
    v16h a;
#pragma unroll
    for (int i = 0; i < 8; ++i) {
      a[i]     = alo[i];
      a[8 + i] = ahi[i];
    }
    // B fragment: column nbase+nl, 16 contiguous halves of K-chunk (kt*2+grp)
    const int bc = kt * 2 + grp;
    v16h b = *(const v16h*)(Wp + (size_t)bc * (kD * 16) +
                            (size_t)(nbase + nl) * 16);
    acc = __builtin_amdgcn_wmma_f32_16x16x32_f16(false, a, false, b, (short)0,
                                                 acc, false, false);
  }

  const float bn = bias[nbase + nl];
#pragma unroll
  for (int i = 0; i < 8; ++i) {
    // C layout: VGPR i -> row i (lanes 0-15) / row 8+i (lanes 16-31)
    float cv = acc[i] + bn;
    if (relu) cv = fmaxf(cv, 0.0f);
    C[(size_t)(mbase + grp * 8 + i) * kD + (nbase + nl)] = cv;
  }
}

// ---------------- edge scores + BigBird mask + segment max -----------------
// one wave32 per edge; float4 per lane covers the 128-dim dot product.
__global__ __launch_bounds__(256) void k_edge_alpha(
    const float* __restrict__ q, const float* __restrict__ k,
    const long long* __restrict__ src, const long long* __restrict__ dst,
    float* __restrict__ alpha, unsigned* __restrict__ mEnc, unsigned rk0,
    unsigned rk1, int E) {
  const int lane = threadIdx.x & 31;
  const int e    = blockIdx.x * 8 + (threadIdx.x >> 5);
  if (e >= E) return;
  const long long sN = src[e];
  const long long dN = dst[e];
  v4f qv = *(const v4f*)(q + (size_t)dN * kD + lane * 4);
  v4f kv = *(const v4f*)(k + (size_t)sN * kD + lane * 4);
  float p = qv[0] * kv[0] + qv[1] * kv[1] + qv[2] * kv[2] + qv[3] * kv[3];
#pragma unroll
  for (int off = 16; off > 0; off >>= 1) p += __shfl_xor(p, off, 32);

  if (lane == 0) {
    float a = p * 0.08838834764831845f;  // 1/sqrt(128)
    long long dd = sN > dN ? sN - dN : dN - sN;
    if (dd > 64) {  // outside local BLOCK window -> random 10% keep
      const unsigned half = (unsigned)(E >> 1);
      const unsigned ue   = (unsigned)e;
      unsigned o0, o1, word;
      if (ue < half) {
        tf2x32(rk0, rk1, ue, ue + half, &o0, &o1);
        word = o0;
      } else {
        tf2x32(rk0, rk1, ue - half, ue, &o0, &o1);
        word = o1;
      }
      float u = __uint_as_float((word >> 9) | 0x3F800000u) - 1.0f;
      if (u > 0.1f) a = -1.0e9f;  // masked out
    }
    alpha[e] = a;
    atomicMax(mEnc + (int)dN, fenc(a));
  }
}

// ---------------- exp(alpha - m[dst]) and segment sum ----------------------
__global__ void k_edge_exp(float* __restrict__ alpha,
                           const long long* __restrict__ dst,
                           const unsigned* __restrict__ mEnc,
                           float* __restrict__ s, int E) {
  int e = blockIdx.x * blockDim.x + threadIdx.x;
  if (e >= E) return;
  int dN   = (int)dst[e];
  float ex = expf(alpha[e] - fdec(mEnc[dN]));
  alpha[e] = ex;  // reuse buffer for exp values
  atomicAdd(s + dN, ex);
}

// ---------------- weighted scatter of v[src] into out[dst] -----------------
__global__ __launch_bounds__(256) void k_edge_scatter(
    const float* __restrict__ ew, const float* __restrict__ s,
    const float* __restrict__ v, const long long* __restrict__ src,
    const long long* __restrict__ dst, float* __restrict__ out, int E) {
  const int lane = threadIdx.x & 31;
  const int e    = blockIdx.x * 8 + (threadIdx.x >> 5);
  if (e >= E) return;
  const long long sN = src[e];
  const long long dN = dst[e];
  const float w = ew[e] / s[(int)dN];  // broadcast loads across the wave
  v4f vv = *(const v4f*)(v + (size_t)sN * kD + lane * 4);
  float* o = out + (size_t)dN * kD + lane * 4;
  atomicAdd(o + 0, vv[0] * w);
  atomicAdd(o + 1, vv[1] * w);
  atomicAdd(o + 2, vv[2] * w);
  atomicAdd(o + 3, vv[3] * w);
}

// ---------------------------------------------------------------------------
extern "C" void kernel_launch(void* const* d_in, const int* in_sizes, int n_in,
                              void* d_out, int out_size, void* d_ws,
                              size_t ws_size, hipStream_t stream) {
  // inputs: x, edge_index, Wq, bq, Wk, bk, Wv, bv, Wo, bo
  const float*      x  = (const float*)d_in[0];
  const long long*  ei = (const long long*)d_in[1];
  const float* Wq = (const float*)d_in[2];
  const float* bq = (const float*)d_in[3];
  const float* Wk = (const float*)d_in[4];
  const float* bk = (const float*)d_in[5];
  const float* Wv = (const float*)d_in[6];
  const float* bv = (const float*)d_in[7];
  const float* Wo = (const float*)d_in[8];
  const float* bo = (const float*)d_in[9];

  const int N  = in_sizes[0] / kD;        // 40000
  const int E  = in_sizes[1] / 2;         // 640000
  const int L  = in_sizes[2] / kDD;       // 3
  const long long ND = (long long)N * kD;

  const long long* src = ei;
  const long long* dst = ei + E;

  // ---- workspace carve-up (256B aligned) ----
  char* base = (char*)d_ws;
  size_t off = 0;
  auto carve = [&](size_t bytes) -> char* {
    char* p = base + off;
    off = (off + bytes + 255) & ~(size_t)255;
    return p;
  };
  _Float16* h16   = (_Float16*)carve(ND * sizeof(_Float16));   // also ao16
  float*    qkv32 = (float*)carve(3 * ND * sizeof(float));
  float*    q32 = qkv32, *k32 = qkv32 + ND, *v32 = qkv32 + 2 * ND;
  float*    attn32 = (float*)carve(ND * sizeof(float));
  float*    h32    = (float*)carve(ND * sizeof(float));
  float*    alpha  = (float*)carve((size_t)E * sizeof(float));
  unsigned* mEnc   = (unsigned*)carve((size_t)N * sizeof(unsigned));
  float*    sSum   = (float*)carve((size_t)N * sizeof(float));
  _Float16* Wq16 = (_Float16*)carve((size_t)L * kDD * sizeof(_Float16));
  _Float16* Wk16 = (_Float16*)carve((size_t)L * kDD * sizeof(_Float16));
  _Float16* Wv16 = (_Float16*)carve((size_t)L * kDD * sizeof(_Float16));
  _Float16* Wo16 = (_Float16*)carve((size_t)L * kDD * sizeof(_Float16));
  (void)ws_size;

  // ---- pack all weights to f16 fragment layout ----
  const int wtot = L * kDD;
  const int wblk = (wtot + 255) / 256;
  k_pack_w<<<wblk, 256, 0, stream>>>(Wq, Wq16, wtot);
  k_pack_w<<<wblk, 256, 0, stream>>>(Wk, Wk16, wtot);
  k_pack_w<<<wblk, 256, 0, stream>>>(Wv, Wv16, wtot);
  k_pack_w<<<wblk, 256, 0, stream>>>(Wo, Wo16, wtot);

  const int nd4    = (int)(ND / 4);
  const int cvtBlk = (nd4 + 255) / 256;
  const int mTiles = N / 16;                // 2500
  const int eWaves = (E + 7) / 8;           // one wave per edge
  const int eThr   = (E + 255) / 256;
  const int nBlk   = (N + 255) / 256;

  for (int l = 0; l < L; ++l) {
    const float* hin = (l == 0) ? x : h32;

    // h -> f16
    k_f32_to_f16<<<cvtBlk, 256, 0, stream>>>(hin, h16, nd4);

    // q,k,v projections (WMMA)
    k_gemm<<<mTiles, 256, 0, stream>>>(h16, Wq16 + (size_t)l * kDD,
                                       bq + (size_t)l * kD, q32, N, 0);
    k_gemm<<<mTiles, 256, 0, stream>>>(h16, Wk16 + (size_t)l * kDD,
                                       bk + (size_t)l * kD, k32, N, 0);
    k_gemm<<<mTiles, 256, 0, stream>>>(h16, Wv16 + (size_t)l * kDD,
                                       bv + (size_t)l * kD, v32, N, 0);

    // segment-softmax state
    k_init_ms<<<nBlk, 256, 0, stream>>>(mEnc, sSum, N);
    k_zero_f32<<<cvtBlk, 256, 0, stream>>>(attn32, nd4);

    // per-layer threefry key: fold_in(key(42), l)
    unsigned rk0, rk1;
    tf2x32(0u, 42u, 0u, (unsigned)l, &rk0, &rk1);

    k_edge_alpha<<<eWaves, 256, 0, stream>>>(q32, k32, src, dst, alpha, mEnc,
                                             rk0, rk1, E);
    k_edge_exp<<<eThr, 256, 0, stream>>>(alpha, dst, mEnc, sSum, E);
    k_edge_scatter<<<eWaves, 256, 0, stream>>>(alpha, sSum, v32, src, dst,
                                               attn32, E);

    // attn -> f16, then output projection (+ReLU on non-final layers)
    k_f32_to_f16<<<cvtBlk, 256, 0, stream>>>(attn32, h16, nd4);
    float* outp = (l == L - 1) ? (float*)d_out : h32;
    k_gemm<<<mTiles, 256, 0, stream>>>(h16, Wo16 + (size_t)l * kDD,
                                       bo + (size_t)l * kD, outp, N,
                                       (l < L - 1) ? 1 : 0);
  }
}
